// STAR_6227702579401
// MI455X (gfx1250) — compile-verified, matrix operand order checked by reading, not run on hardware
//
#include <hip/hip_runtime.h>

// ---------------------------------------------------------------------------
// STAR multi-domain MLP, fully fused for gfx1250 (MI455X), wave32 + WMMA bf16.
// M=32 row tiles: each packed-B fragment feeds two v_wmma ops (halves L2 BW).
// ---------------------------------------------------------------------------

#define BATCH   65536
#define F_FEAT  16
#define VOCAB   10000
#define EMB     16
#define IN_DIM  256
#define NDOM    5
#define MROWS   32
#define PMAX    (BATCH + NDOM * MROWS)   // 65696 = 32 * 2053

typedef __attribute__((ext_vector_type(16))) __bf16 v16bf;
typedef __attribute__((ext_vector_type(8)))  float  v8f;

union BF16x16 {
    v16bf v;
    unsigned short h[16];
    uint4 q[2];
};

__device__ __forceinline__ unsigned short f2bf(float f) {
    union { float f; unsigned u; } x; x.f = f;
    unsigned r = x.u + 0x7FFFu + ((x.u >> 16) & 1u);   // RNE
    return (unsigned short)(r >> 16);
}
__device__ __forceinline__ float bf2f(unsigned short h) {
    union { unsigned u; float f; } x; x.u = ((unsigned)h) << 16;
    return x.f;
}

// ---------------------------------------------------------------------------
// Packed weight layout: Wp[(n * (K/32) + c) * 32 + kk] = bf16(W[c*32+kk, n])
// -> a lane's 32x16 B-fragment = two contiguous global_load_b128.
// ---------------------------------------------------------------------------
__global__ void star_pack_k(const float* __restrict__ W, const float* __restrict__ Wm,
                            unsigned short* __restrict__ Wp, int K, int N) {
    const int kc32 = K >> 5;
    const int total = N * kc32;
    int i = blockIdx.x * blockDim.x + threadIdx.x;
    if (i >= total) return;
    const int c = i % kc32;
    const int n = i / kc32;
    const float* src = W + (size_t)(c << 5) * N + n;
    unsigned short* dst = Wp + ((size_t)i << 5);
    if (Wm) {
        const float* srcm = Wm + (size_t)(c << 5) * N + n;
        for (int kk = 0; kk < 32; ++kk)
            dst[kk] = f2bf(src[(size_t)kk * N] * srcm[(size_t)kk * N]);
    } else {
        for (int kk = 0; kk < 32; ++kk)
            dst[kk] = f2bf(src[(size_t)kk * N]);
    }
}

// ---------------------------------------------------------------------------
// 32-row GEMM phase: C(32xN) = A(32xK bf16, LDS) @ Wp(packed bf16, L2).
// 8 waves; wave owns TPW 16-col tiles; 2 row sub-tiles share each B fragment.
// DS = destination row stride (elements). Scalar guards only (EXEC all-ones
// at every v_wmma). Optional bias2, LayerNorm+ReLU, residual.
// ---------------------------------------------------------------------------
template<int TPW, int K, int N, int DS, bool B2, bool LN, bool RES>
__device__ __forceinline__ void wmma_gemm(
    const unsigned short* __restrict__ As,
    const unsigned short* __restrict__ Wp,
    const float* __restrict__ b1, const float* __restrict__ b2,
    const float* __restrict__ g,  const float* __restrict__ beta,
    const unsigned short* __restrict__ resid,
    unsigned short* __restrict__ dst,
    float* red_s, float* red_q, float* stats)
{
    constexpr int NT = N >> 4;
    constexpr bool FULL = (NT == TPW * 8);
    const int tid  = threadIdx.x;
    const int wave = __builtin_amdgcn_readfirstlane(tid >> 5);   // scalar
    const int lane = tid & 31;
    const int half = lane >> 4;
    const int lrow = lane & 15;
    const bool active = FULL || (wave * TPW < NT);               // scalar branch

    v8f acc[2][TPW];
#pragma unroll
    for (int s = 0; s < 2; ++s)
#pragma unroll
        for (int i = 0; i < TPW; ++i)
            acc[s][i] = v8f{0.f, 0.f, 0.f, 0.f, 0.f, 0.f, 0.f, 0.f};

    if (active) {
        for (int kc = 0; kc < K; kc += 32) {
            BF16x16 a0, a1;
            a0.q[0] = *(const uint4*)(As + lrow * K + kc + half * 8);
            a0.q[1] = *(const uint4*)(As + lrow * K + kc + 16 + half * 8);
            a1.q[0] = *(const uint4*)(As + (16 + lrow) * K + kc + half * 8);
            a1.q[1] = *(const uint4*)(As + (16 + lrow) * K + kc + 16 + half * 8);
#pragma unroll
            for (int i = 0; i < TPW; ++i) {
                const int n = ((wave * TPW + i) << 4) + lrow;
                const unsigned short* bpp =
                    Wp + (((size_t)n * (K >> 5) + (kc >> 5)) << 5);
                BF16x16 b;
                b.q[0] = *(const uint4*)(bpp + half * 8);
                b.q[1] = *(const uint4*)(bpp + 16 + half * 8);
                acc[0][i] = __builtin_amdgcn_wmma_f32_16x16x32_bf16(
                    false, a0.v, false, b.v, (short)0, acc[0][i], false, false);
                acc[1][i] = __builtin_amdgcn_wmma_f32_16x16x32_bf16(
                    false, a1.v, false, b.v, (short)0, acc[1][i], false, false);
            }
        }
        // bias
#pragma unroll
        for (int i = 0; i < TPW; ++i) {
            const int n = ((wave * TPW + i) << 4) + lrow;
            float bb = b1[n];
            if (B2) bb += b2[n];
#pragma unroll
            for (int s = 0; s < 2; ++s)
#pragma unroll
                for (int r = 0; r < 8; ++r) acc[s][i][r] += bb;
        }
    }

    if (LN) { if (tid < 32) { red_s[tid] = 0.f; red_q[tid] = 0.f; } }
    __syncthreads();   // all A reads done (dst may alias As); red zeroed

    if (LN) {
        if (active) {
            float rs[2][8], rq[2][8];
#pragma unroll
            for (int s = 0; s < 2; ++s)
#pragma unroll
                for (int r = 0; r < 8; ++r) { rs[s][r] = 0.f; rq[s][r] = 0.f; }
#pragma unroll
            for (int i = 0; i < TPW; ++i)
#pragma unroll
                for (int s = 0; s < 2; ++s)
#pragma unroll
                    for (int r = 0; r < 8; ++r) {
                        float v = acc[s][i][r];
                        rs[s][r] += v; rq[s][r] += v * v;
                    }
#pragma unroll
            for (int s = 0; s < 2; ++s)
#pragma unroll
                for (int r = 0; r < 8; ++r) {
                    atomicAdd(&red_s[s * 16 + r + 8 * half], rs[s][r]);
                    atomicAdd(&red_q[s * 16 + r + 8 * half], rq[s][r]);
                }
        }
        __syncthreads();
        if (tid < 32) {
            float m   = red_s[tid] * (1.f / (float)N);
            float var = red_q[tid] * (1.f / (float)N) - m * m;
            stats[tid]      = m;
            stats[32 + tid] = rsqrtf(var + 1e-5f);
        }
        __syncthreads();
        if (active) {
#pragma unroll
            for (int i = 0; i < TPW; ++i) {
                const int n  = ((wave * TPW + i) << 4) + lrow;
                const float gg = g[n], b2v = beta[n];
#pragma unroll
                for (int s = 0; s < 2; ++s)
#pragma unroll
                    for (int r = 0; r < 8; ++r) {
                        const int row = s * 16 + r + 8 * half;
                        float v = (acc[s][i][r] - stats[row]) * stats[32 + row] * gg + b2v;
                        v = fmaxf(v, 0.f);
                        if (RES) v += bf2f(resid[row * DS + n]);
                        dst[row * DS + n] = f2bf(v);
                    }
            }
        }
    } else {
        if (active) {
#pragma unroll
            for (int i = 0; i < TPW; ++i) {
                const int n = ((wave * TPW + i) << 4) + lrow;
#pragma unroll
                for (int s = 0; s < 2; ++s)
#pragma unroll
                    for (int r = 0; r < 8; ++r)
                        dst[(s * 16 + r + 8 * half) * DS + n] = f2bf(acc[s][i][r]);
            }
        }
    }
    __syncthreads();
}

// ---------------------------------------------------------------------------
// Setup kernels: bucket rows by domain (padded-to-32 segments) on device.
// ---------------------------------------------------------------------------
__global__ void star_init_k(int* wsI, int* perm, int pmax) {
    int i = blockIdx.x * blockDim.x + threadIdx.x;
    if (i < 24) wsI[i] = 0;
    if (i < pmax) perm[i] = -1;
}
__global__ void star_count_k(const int* __restrict__ xdom, int* cnt) {
    int i = blockIdx.x * blockDim.x + threadIdx.x;
    if (i < BATCH) atomicAdd(&cnt[xdom[i]], 1);
}
__global__ void star_scan_k(const int* __restrict__ cnt, int* off) {
    if (threadIdx.x == 0 && blockIdx.x == 0) {
        int o = 0;
        for (int d = 0; d < NDOM; ++d) { off[d] = o; o += (cnt[d] + 31) & ~31; }
        off[NDOM] = o;
    }
}
__global__ void star_fill_k(const int* __restrict__ xdom, const int* __restrict__ off,
                            int* cur, int* perm) {
    int i = blockIdx.x * blockDim.x + threadIdx.x;
    if (i < BATCH) {
        int d = xdom[i];
        int p = off[d] + atomicAdd(&cur[d], 1);
        perm[p] = i;
    }
}

// ---------------------------------------------------------------------------
// Megakernel: one 32-row tile end-to-end. LDS overlapped by lifetime (~113 KB
// of the 320 KB WGP LDS).
// ---------------------------------------------------------------------------
#define SM_H1    0         // 32x1024 bf16 (64 KB) ; later H2/H4/H6
#define SM_H2    0         // 32x512 (32 KB)
#define SM_H3    32768     // 32x256 (16 KB) ; later H5
#define SM_H4    0
#define SM_H5    32768
#define SM_H6    0
#define SM_SKIP  65536     // 32x512 bf16 (32 KB)
#define SM_AS0   98304     // 32x256 bf16 (16 KB)
#define SM_RED   114688    // red_s[32], red_q[32], stats[64]
#define SM_BYTES 115200

__global__ __launch_bounds__(256) void star_mega_k(
    const int*   __restrict__ x,   const float* __restrict__ tables,
    const unsigned short* __restrict__ pSkip, const unsigned short* __restrict__ pDom,
    const unsigned short* __restrict__ pStar, const unsigned short* __restrict__ pM0,
    const unsigned short* __restrict__ pM1,   const unsigned short* __restrict__ pM2,
    const unsigned short* __restrict__ pH1,
    const float* __restrict__ skip_b, const float* __restrict__ skip_g,
    const float* __restrict__ skip_be,
    const float* __restrict__ shared_b, const float* __restrict__ slot_b,
    const float* __restrict__ star_b, const float* __restrict__ star_g,
    const float* __restrict__ star_be,
    const float* __restrict__ mb0, const float* __restrict__ mg0,
    const float* __restrict__ mbe0,
    const float* __restrict__ mb1, const float* __restrict__ mg1,
    const float* __restrict__ mbe1,
    const float* __restrict__ mb2, const float* __restrict__ mg2,
    const float* __restrict__ mbe2,
    const float* __restrict__ hb1, const float* __restrict__ hg,
    const float* __restrict__ hbe,
    const float* __restrict__ hW2, const float* __restrict__ hb2,
    const int* __restrict__ off, const int* __restrict__ perm,
    float* __restrict__ out)
{
    __shared__ __align__(16) unsigned char smem[SM_BYTES];
    unsigned short* H1  = (unsigned short*)(smem + SM_H1);
    unsigned short* H2  = (unsigned short*)(smem + SM_H2);
    unsigned short* H3  = (unsigned short*)(smem + SM_H3);
    unsigned short* H4  = (unsigned short*)(smem + SM_H4);
    unsigned short* H5  = (unsigned short*)(smem + SM_H5);
    unsigned short* H6  = (unsigned short*)(smem + SM_H6);
    unsigned short* SK  = (unsigned short*)(smem + SM_SKIP);
    unsigned short* AS0 = (unsigned short*)(smem + SM_AS0);
    float* red_s = (float*)(smem + SM_RED);
    float* red_q = red_s + 32;
    float* stats = red_s + 64;

    const int tid  = threadIdx.x;
    const int base = blockIdx.x * MROWS;
    if (base >= off[NDOM]) return;      // uniform early-exit past padded end

    const int d = (base >= off[1]) + (base >= off[2]) + (base >= off[3]) + (base >= off[4]);

    // ---- embedding gather: 512 (row,feature) tasks, 2 per thread ---------
#pragma unroll
    for (int tsk = 0; tsk < 2; ++tsk) {
        const int task = tid + tsk * 256;
        const int r = task >> 4, f = task & 15;
        const int orig = perm[base + r];
        unsigned short tmp[16];
        if (orig >= 0) {
            const int idx = x[f * BATCH + orig];
            const float4* tp = (const float4*)(tables + ((size_t)f * VOCAB + idx) * EMB);
#pragma unroll
            for (int q = 0; q < 4; ++q) {
                float4 v = tp[q];
                tmp[q * 4 + 0] = f2bf(v.x); tmp[q * 4 + 1] = f2bf(v.y);
                tmp[q * 4 + 2] = f2bf(v.z); tmp[q * 4 + 3] = f2bf(v.w);
            }
        } else {
#pragma unroll
            for (int q = 0; q < 16; ++q) tmp[q] = 0;
        }
#pragma unroll
        for (int q = 0; q < 16; ++q) AS0[r * IN_DIM + f * 16 + q] = tmp[q];
    }
    __syncthreads();

    // ---- skip branch: 256 -> 512, LN+ReLU --------------------------------
    wmma_gemm<4, 256, 512, 512, false, true, false>(
        AS0, pSkip, skip_b, nullptr, skip_g, skip_be, nullptr, SK,
        red_s, red_q, stats);

    // ---- domain branch: 256 -> 1024 (pre-fused slot*shared), two halves --
    const unsigned short* pDomD = pDom + (size_t)d * IN_DIM * 1024;
    wmma_gemm<4, 256, 512, 1024, true, false, false>(
        AS0, pDomD, slot_b + d * 1024, shared_b,
        nullptr, nullptr, nullptr, H1, red_s, red_q, stats);
    wmma_gemm<4, 256, 512, 1024, true, false, false>(
        AS0, pDomD + (size_t)512 * (IN_DIM >> 5) * 32,
        slot_b + d * 1024 + 512, shared_b + 512,
        nullptr, nullptr, nullptr, H1 + 512, red_s, red_q, stats);

    // ---- star: 1024 -> 512, LN+ReLU + skip residual ----------------------
    wmma_gemm<4, 1024, 512, 512, false, true, true>(
        H1, pStar, star_b, nullptr, star_g, star_be, SK, H2,
        red_s, red_q, stats);

    // ---- main tower ------------------------------------------------------
    wmma_gemm<2, 512, 256, 256, false, true, false>(
        H2, pM0, mb0, nullptr, mg0, mbe0, nullptr, H3, red_s, red_q, stats);
    wmma_gemm<2, 256, 256, 256, false, true, false>(
        H3, pM1, mb1, nullptr, mg1, mbe1, nullptr, H4, red_s, red_q, stats);
    wmma_gemm<1, 256, 64, 64, false, true, false>(
        H4, pM2, mb2, nullptr, mg2, mbe2, nullptr, H5, red_s, red_q, stats);

    // ---- head layer 1: 64 -> 64, LN+ReLU ---------------------------------
    wmma_gemm<1, 64, 64, 64, false, true, false>(
        H5, pH1, hb1, nullptr, hg, hbe, nullptr, H6, red_s, red_q, stats);

    // ---- head layer 2: 64 -> 1, sigmoid, scatter -------------------------
    if (tid < 32) {
        const int orig = perm[base + tid];
        if (orig >= 0) {
            float s = hb2[0];
#pragma unroll 8
            for (int k = 0; k < 64; ++k) s += bf2f(H6[tid * 64 + k]) * hW2[k];
            out[orig] = 1.f / (1.f + __expf(-s));
        }
    }
}

// ---------------------------------------------------------------------------
// Workspace layout (bytes)
// ---------------------------------------------------------------------------
#define WS_INTS   0                         // cnt[8] cur[8] off[8] perm[PMAX]
#define WS_PSKIP  263168                    // (32+PMAX)*4=262912 -> align 256
#define WS_PDOM   (WS_PSKIP + 262144)       // 5x256x1024 bf16 (2621440 B)
#define WS_PSTAR  (WS_PDOM + 2621440)       // 1024x512  bf16  (1048576 B)
#define WS_PM0    (WS_PSTAR + 1048576)      // 512x256   bf16  (262144 B)
#define WS_PM1    (WS_PM0 + 262144)         // 256x256   bf16  (131072 B)
#define WS_PM2    (WS_PM1 + 131072)         // 256x64    bf16  (32768 B)
#define WS_PH1    (WS_PM2 + 32768)          // 64x64     bf16  (8192 B)

extern "C" void kernel_launch(void* const* d_in, const int* in_sizes, int n_in,
                              void* d_out, int out_size, void* d_ws, size_t ws_size,
                              hipStream_t stream)
{
    (void)in_sizes; (void)n_in; (void)out_size; (void)ws_size;

    const int*   x        = (const int*)  d_in[0];
    const float* tables   = (const float*)d_in[1];
    const float* skip_W   = (const float*)d_in[2];
    const float* skip_b   = (const float*)d_in[3];
    const float* skip_g   = (const float*)d_in[4];
    const float* skip_be  = (const float*)d_in[5];
    const float* shared_W = (const float*)d_in[6];
    const float* shared_b = (const float*)d_in[7];
    const float* slot_W   = (const float*)d_in[8];
    const float* slot_b   = (const float*)d_in[9];
    const float* star_W   = (const float*)d_in[10];
    const float* star_b   = (const float*)d_in[11];
    const float* star_g   = (const float*)d_in[12];
    const float* star_be  = (const float*)d_in[13];
    const float* mW0 = (const float*)d_in[14]; const float* mb0 = (const float*)d_in[15];
    const float* mg0 = (const float*)d_in[16]; const float* mbe0= (const float*)d_in[17];
    const float* mW1 = (const float*)d_in[18]; const float* mb1 = (const float*)d_in[19];
    const float* mg1 = (const float*)d_in[20]; const float* mbe1= (const float*)d_in[21];
    const float* mW2 = (const float*)d_in[22]; const float* mb2 = (const float*)d_in[23];
    const float* mg2 = (const float*)d_in[24]; const float* mbe2= (const float*)d_in[25];
    const float* hW1 = (const float*)d_in[26]; const float* hb1 = (const float*)d_in[27];
    const float* hg  = (const float*)d_in[28]; const float* hbe = (const float*)d_in[29];
    const float* hW2 = (const float*)d_in[30]; const float* hb2 = (const float*)d_in[31];

    unsigned char* ws = (unsigned char*)d_ws;
    int* wsI  = (int*)(ws + WS_INTS);
    int* cnt  = wsI;
    int* cur  = wsI + 8;
    int* off  = wsI + 16;
    int* perm = wsI + 32;
    unsigned short* pSkip = (unsigned short*)(ws + WS_PSKIP);
    unsigned short* pDom  = (unsigned short*)(ws + WS_PDOM);
    unsigned short* pStar = (unsigned short*)(ws + WS_PSTAR);
    unsigned short* pM0   = (unsigned short*)(ws + WS_PM0);
    unsigned short* pM1   = (unsigned short*)(ws + WS_PM1);
    unsigned short* pM2   = (unsigned short*)(ws + WS_PM2);
    unsigned short* pH1   = (unsigned short*)(ws + WS_PH1);

    const int* xdom = x + (F_FEAT - 1) * BATCH;

    // ---- weight pre-pack (fragment-native bf16) --------------------------
    auto packN = [](int K, int N) { return N * (K >> 5); };
    star_pack_k<<<(packN(256,512)+255)/256, 256, 0, stream>>>(skip_W, nullptr, pSkip, 256, 512);
    for (int d = 0; d < NDOM; ++d)
        star_pack_k<<<(packN(256,1024)+255)/256, 256, 0, stream>>>(
            slot_W + (size_t)d * IN_DIM * 1024, shared_W,
            pDom + (size_t)d * IN_DIM * 1024, 256, 1024);
    star_pack_k<<<(packN(1024,512)+255)/256, 256, 0, stream>>>(star_W, nullptr, pStar, 1024, 512);
    star_pack_k<<<(packN(512,256)+255)/256, 256, 0, stream>>>(mW0, nullptr, pM0, 512, 256);
    star_pack_k<<<(packN(256,256)+255)/256, 256, 0, stream>>>(mW1, nullptr, pM1, 256, 256);
    star_pack_k<<<(packN(256,64)+255)/256, 256, 0, stream>>>(mW2, nullptr, pM2, 256, 64);
    star_pack_k<<<(packN(64,64)+255)/256, 256, 0, stream>>>(hW1, nullptr, pH1, 64, 64);

    // ---- domain bucketing ------------------------------------------------
    star_init_k <<<(PMAX + 255) / 256, 256, 0, stream>>>(wsI, perm, PMAX);
    star_count_k<<<BATCH / 256, 256, 0, stream>>>(xdom, cnt);
    star_scan_k <<<1, 32, 0, stream>>>(cnt, off);
    star_fill_k <<<BATCH / 256, 256, 0, stream>>>(xdom, off, cur, perm);

    // ---- fused network ---------------------------------------------------
    star_mega_k<<<PMAX / MROWS, 256, 0, stream>>>(
        x, tables,
        pSkip, pDom, pStar, pM0, pM1, pM2, pH1,
        skip_b, skip_g, skip_be,
        shared_b, slot_b, star_b, star_g, star_be,
        mb0, mg0, mbe0, mb1, mg1, mbe1, mb2, mg2, mbe2,
        hb1, hg, hbe, hW2, hb2,
        off, perm, (float*)d_out);
}
